// MultiHeadAttention_40424232190302
// MI455X (gfx1250) — matrix operand order repaired and checked by reading, not compiled, not run on hardware
//
#include <hip/hip_runtime.h>

// ---------------------------------------------------------------------------
// MultiHeadAttention forward (B=2, L=4096, H=1024, NH=16, HD=64, RoPE, softmax)
// MI455X / gfx1250: all GEMMs via v_wmma_f32_16x16x32_bf16, flash-attention
// online softmax (scores never materialized), LDS staging for K/V/P tiles.
// ---------------------------------------------------------------------------

typedef __attribute__((ext_vector_type(16))) __bf16 v16bf;
typedef __attribute__((ext_vector_type(8)))  __bf16 v8bf;
typedef __attribute__((ext_vector_type(8)))  float  v8f;

constexpr int kB  = 2;
constexpr int kL  = 4096;
constexpr int kH  = 1024;
constexpr int kNH = 16;
constexpr int kHD = 64;

// ---- helpers --------------------------------------------------------------

static __device__ __forceinline__ __bf16 f2bf(float f) {
  union { float f; unsigned u; } a; a.f = f;
  unsigned r = a.u + 0x7FFFu + ((a.u >> 16) & 1u);   // round-to-nearest-even
  union { unsigned short s; __bf16 b; } o;
  o.s = (unsigned short)(r >> 16);
  return o.b;
}

static __device__ __forceinline__ v8f zero8() {
  v8f z;
  #pragma unroll
  for (int i = 0; i < 8; ++i) z[i] = 0.0f;
  return z;
}

// Build a 16-element bf16 fragment from two 16B (b128) loads.
static __device__ __forceinline__ v16bf ldfrag(const __bf16* p0, const __bf16* p1) {
  v8bf a = *(const v8bf*)p0;
  v8bf b = *(const v8bf*)p1;
  return __builtin_shufflevector(a, b, 0, 1, 2, 3, 4, 5, 6, 7,
                                 8, 9, 10, 11, 12, 13, 14, 15);
}

static __device__ __forceinline__ v8f wmma_bf16(v16bf a, v16bf b, v8f c) {
#if defined(__gfx1250__)
  return __builtin_amdgcn_wmma_f32_16x16x32_bf16(
      /*neg_a=*/false, a, /*neg_b=*/false, b,
      /*c_mod=*/(short)0, c, /*reuse_a=*/false, /*reuse_b=*/false);
#else
  (void)a; (void)b;
  return c;
#endif
}

static __device__ __forceinline__ void sched_fence() {
#if defined(__gfx1250__)
  __builtin_amdgcn_sched_barrier(0);
#endif
}

// ---- shared GEMM mainloop: C[32x64] per wave, K = 1024 -------------------
// A: [M x 1024] row-major bf16.  W: [N x 1024] row-major bf16 (Linear weight,
// out = A @ W^T).  Per-lane fragment addressing uses a contiguous-K order (not AMD's
// table order), same for A and B: A lane = M row, elements K=(hi?16:0)+0..15; B lane = N
// col, elements K=(hi?16:0)+0..15 (all contiguous-K -> b128 loads).
static __device__ __forceinline__ void gemm_mainloop(
    const __bf16* __restrict__ A, const __bf16* __restrict__ W,
    int m0, int n0, int lane, v8f acc[2][4]) {
  const int l15 = lane & 15;
  const int hi  = (lane >> 4) & 1;
  const __bf16* aRow0 = A + (size_t)(m0 + l15) * kH;
  const __bf16* aRow1 = A + (size_t)(m0 + 16 + l15) * kH;
  const __bf16* wRow0 = W + (size_t)(n0 + 0 * 16 + l15) * kH;
  const __bf16* wRow1 = W + (size_t)(n0 + 1 * 16 + l15) * kH;
  const __bf16* wRow2 = W + (size_t)(n0 + 2 * 16 + l15) * kH;
  const __bf16* wRow3 = W + (size_t)(n0 + 3 * 16 + l15) * kH;

  #pragma unroll
  for (int i = 0; i < 2; ++i)
    #pragma unroll
    for (int j = 0; j < 4; ++j) acc[i][j] = zero8();

  for (int k = 0; k < kH; k += 32) {
    const int ka = k + hi * 16;   // A fragment element base
    const int kb = k + hi * 16;   // B fragment element base
    v16bf a0 = ldfrag(aRow0 + ka, aRow0 + ka + 8);
    v16bf a1 = ldfrag(aRow1 + ka, aRow1 + ka + 8);
    v16bf b0 = ldfrag(wRow0 + kb, wRow0 + kb + 8);
    v16bf b1 = ldfrag(wRow1 + kb, wRow1 + kb + 8);
    v16bf b2 = ldfrag(wRow2 + kb, wRow2 + kb + 8);
    v16bf b3 = ldfrag(wRow3 + kb, wRow3 + kb + 8);
    acc[0][0] = wmma_bf16(a0, b0, acc[0][0]);
    acc[1][0] = wmma_bf16(a1, b0, acc[1][0]);
    acc[0][1] = wmma_bf16(a0, b1, acc[0][1]);
    acc[1][1] = wmma_bf16(a1, b1, acc[1][1]);
    acc[0][2] = wmma_bf16(a0, b2, acc[0][2]);
    acc[1][2] = wmma_bf16(a1, b2, acc[1][2]);
    acc[0][3] = wmma_bf16(a0, b3, acc[0][3]);
    acc[1][3] = wmma_bf16(a1, b3, acc[1][3]);
  }
}

// ---- kernel 0: fp32 -> bf16 conversion ------------------------------------

struct __align__(8) Bf4 { __bf16 x, y, z, w; };

__global__ __launch_bounds__(256) void cvt_kernel(
    const float* __restrict__ in, __bf16* __restrict__ out, int n4) {
  int i = blockIdx.x * blockDim.x + threadIdx.x;
  if (i < n4) {
    float4 v = ((const float4*)in)[i];
    Bf4 o;
    o.x = f2bf(v.x); o.y = f2bf(v.y); o.z = f2bf(v.z); o.w = f2bf(v.w);
    ((Bf4*)out)[i] = o;
  }
}

// ---- kernel 1: fused QKV projection + RoPE --------------------------------
// grid: (M/128, N/128, 3)  block: 256 (8 waves, 4x2 wave grid, 32x64/wave)
// outputs Q (RoPE, pre-scaled by 1/sqrt(HD)), K (RoPE), V in [B,NH,L,HD] bf16.

__global__ __launch_bounds__(256) void qkv_proj_kernel(
    const __bf16* __restrict__ X,
    const __bf16* __restrict__ Wq, const __bf16* __restrict__ Wk,
    const __bf16* __restrict__ Wv,
    __bf16* __restrict__ Qb, __bf16* __restrict__ Kb, __bf16* __restrict__ Vb) {
  const int z = blockIdx.z;
  const __bf16* W = (z == 0) ? Wq : (z == 1) ? Wk : Wv;
  __bf16* Out     = (z == 0) ? Qb : (z == 1) ? Kb : Vb;

  const int lane = threadIdx.x & 31;
  const int wave = threadIdx.x >> 5;
  const int m0 = blockIdx.x * 128 + (wave & 3) * 32;
  const int n0 = blockIdx.y * 128 + (wave >> 2) * 64;  // head-aligned (64)

  v8f acc[2][4];
  gemm_mainloop(X, W, m0, n0, lane, acc);

  const int l15 = lane & 15, hi = (lane >> 4) & 1;
  const float qscale = (z == 0) ? 0.125f : 1.0f;  // 1/sqrt(64) folded into Q

  if (z < 2) {
    // RoPE: within this wave's 64-wide (one head) tile, pair d and d+32,
    // i.e. acc[.][j] with acc[.][j+2]; same lane, same register element.
    #pragma unroll
    for (int i = 0; i < 2; ++i) {
      #pragma unroll
      for (int j = 0; j < 2; ++j) {
        const int d1 = j * 16 + l15;                    // 0..31
        // inv_freq = 10000^(-d1/32) = exp(-d1 * ln(10000)/32)
        const float invf = __expf(-(float)d1 * (9.210340371976184f / 32.0f));
        #pragma unroll
        for (int r = 0; r < 8; ++r) {
          const int m = m0 + i * 16 + hi * 8 + r;
          const float t = (float)(m & (kL - 1));        // sequence position
          float s, c;
          __sincosf(t * invf, &s, &c);
          const float x1 = acc[i][j][r];
          const float x2 = acc[i][j + 2][r];
          acc[i][j][r]     = x1 * c - x2 * s;
          acc[i][j + 2][r] = x2 * c + x1 * s;
        }
      }
    }
  }

  // store [B, NH, L, HD]
  #pragma unroll
  for (int i = 0; i < 2; ++i)
    #pragma unroll
    for (int j = 0; j < 4; ++j)
      #pragma unroll
      for (int r = 0; r < 8; ++r) {
        const int m = m0 + i * 16 + hi * 8 + r;
        const int n = n0 + j * 16 + l15;
        const int b = m >> 12, l = m & (kL - 1);
        const int h = n >> 6,  d = n & 63;
        const size_t idx = (((size_t)(b * kNH + h) * kL) + l) * kHD + d;
        Out[idx] = f2bf(acc[i][j][r] * qscale);
      }
}

// ---- kernel 2: flash attention -------------------------------------------
// grid: (L/64, B*NH)  block: 128 (4 waves, 16 q-rows each).
// Per 64-key tile: stage K (row-major) and V (transposed) in LDS, S = Q K^T
// via WMMA, +mask, online softmax in registers (shfl reductions over the
// 16-lane halves), P -> LDS -> A-fragments, O += P V via WMMA.

#define LDSTR 72  // LDS row stride in bf16 elements (144B, keeps b128 aligned)

__global__ __launch_bounds__(128) void flash_kernel(
    const __bf16* __restrict__ Qb, const __bf16* __restrict__ Kb,
    const __bf16* __restrict__ Vb, const float* __restrict__ mask,
    __bf16* __restrict__ Ab) {
  __shared__ __align__(16) __bf16 Kt[64 * LDSTR];      // [key][d]
  __shared__ __align__(16) __bf16 Vt[64 * LDSTR];      // [d][key] (transposed)
  __shared__ __align__(16) __bf16 Pt[4][16 * LDSTR];   // per-wave [q][key]

  const int tid  = threadIdx.x;
  const int lane = tid & 31, w = tid >> 5;
  const int l15  = lane & 15, hi = (lane >> 4) & 1;
  const int q0   = blockIdx.x * 64;
  const int bh   = blockIdx.y;
  const int b = bh >> 4, h = bh & 15;

  const __bf16* Qh = Qb + (size_t)bh * kL * kHD;
  const __bf16* Kh = Kb + (size_t)bh * kL * kHD;
  const __bf16* Vh = Vb + (size_t)bh * kL * kHD;
  const float* maskB = mask + (size_t)b * kL * kL;

  // Q A-fragments for this wave's 16 rows (d 0..31 and 32..63), RoPE+scale
  // already applied at projection time.
  const __bf16* qPtr = Qh + (size_t)(q0 + w * 16 + l15) * kHD;
  const v16bf aq0 = ldfrag(qPtr + hi * 16,      qPtr + hi * 16 + 8);
  const v16bf aq1 = ldfrag(qPtr + 32 + hi * 16, qPtr + 32 + hi * 16 + 8);

  v8f accO[4];
  #pragma unroll
  for (int j = 0; j < 4; ++j) accO[j] = zero8();
  float mrow[8], lrow[8];
  #pragma unroll
  for (int r = 0; r < 8; ++r) { mrow[r] = -1e30f; lrow[r] = 0.0f; }

  for (int kt = 0; kt < kL / 64; ++kt) {
    const int k0 = kt * 64;
    // ---- stage K tile (row-major, v8bf = b128 copies) ----
    #pragma unroll
    for (int it = 0; it < 4; ++it) {
      const int idx = tid + it * 128;          // 512 chunks of 8 halves
      const int key = idx >> 3, c = idx & 7;
      *(v8bf*)(Kt + key * LDSTR + c * 8) =
          *(const v8bf*)(Kh + (size_t)(k0 + key) * kHD + c * 8);
    }
    // ---- stage V transposed: Vt[d][key] ----
    #pragma unroll
    for (int it = 0; it < 32; ++it) {
      const int idx = tid + it * 128;          // 4096 elements
      const int d = idx & 63, key = idx >> 6;  // global read coalesced in d
      Vt[d * LDSTR + key] = Vh[(size_t)(k0 + key) * kHD + d];
    }
    __syncthreads();

    // ---- S = Q K^T (already includes 1/sqrt(HD) via Q) ----
    v8f accS[4];
    #pragma unroll
    for (int j = 0; j < 4; ++j) {
      const __bf16* kp = Kt + (j * 16 + l15) * LDSTR;  // B lane = key col
      v16bf b0 = ldfrag(kp + hi * 16,      kp + hi * 16 + 8);       // d 0..31
      v16bf b1 = ldfrag(kp + 32 + hi * 16, kp + 32 + hi * 16 + 8);  // d 32..63
      v8f t = wmma_bf16(aq0, b0, zero8());
      accS[j] = wmma_bf16(aq1, b1, t);
    }

    // ---- +mask, online softmax (stats per register element = q row) ----
    float p[4][8];
    #pragma unroll
    for (int r = 0; r < 8; ++r) {
      const int qg = q0 + w * 16 + hi * 8 + r;
      const float* mrp = maskB + (size_t)qg * kL + k0;
      float s0 = accS[0][r] + mrp[l15];
      float s1 = accS[1][r] + mrp[16 + l15];
      float s2 = accS[2][r] + mrp[32 + l15];
      float s3 = accS[3][r] + mrp[48 + l15];
      float mx = fmaxf(fmaxf(s0, s1), fmaxf(s2, s3));
      #pragma unroll
      for (int off = 1; off <= 8; off <<= 1)
        mx = fmaxf(mx, __shfl_xor(mx, off, 32));     // reduce 16-lane half
      const float mnew  = fmaxf(mrow[r], mx);
      const float alpha = __expf(mrow[r] - mnew);
      mrow[r] = mnew;
      const float p0 = __expf(s0 - mnew), p1 = __expf(s1 - mnew);
      const float p2 = __expf(s2 - mnew), p3 = __expf(s3 - mnew);
      float ls = (p0 + p1) + (p2 + p3);
      #pragma unroll
      for (int off = 1; off <= 8; off <<= 1)
        ls += __shfl_xor(ls, off, 32);
      lrow[r] = lrow[r] * alpha + ls;
      #pragma unroll
      for (int j = 0; j < 4; ++j) accO[j][r] *= alpha;
      p[0][r] = p0; p[1][r] = p1; p[2][r] = p2; p[3][r] = p3;
    }

    // ---- P (C layout) -> per-wave LDS -> A-fragment layout ----
    __bf16* pw = &Pt[w][0];
    #pragma unroll
    for (int j = 0; j < 4; ++j)
      #pragma unroll
      for (int r = 0; r < 8; ++r)
        pw[(hi * 8 + r) * LDSTR + j * 16 + l15] = f2bf(p[j][r]);
    sched_fence();  // same-wave DS ops are in-order; keep compiler ordered too

    // ---- O += P V ----
    #pragma unroll
    for (int c = 0; c < 2; ++c) {                       // key chunks of 32
      const __bf16* pp = pw + l15 * LDSTR + c * 32 + hi * 16;
      v16bf ap = ldfrag(pp, pp + 8);
      #pragma unroll
      for (int j = 0; j < 4; ++j) {                     // d tiles of 16
        const __bf16* vp = Vt + (j * 16 + l15) * LDSTR + c * 32 + hi * 16;
        v16bf bv = ldfrag(vp, vp + 8);
        accO[j] = wmma_bf16(ap, bv, accO[j]);
      }
    }
    __syncthreads();  // protect Kt/Vt before next tile's staging
  }

  // ---- normalize and store attn output as [B*L, H] bf16 (heads concat) ----
  #pragma unroll
  for (int r = 0; r < 8; ++r) {
    const int qg = q0 + w * 16 + hi * 8 + r;
    const float inv = (lrow[r] > 0.0f) ? 1.0f / lrow[r] : 0.0f;
    const size_t base = ((size_t)(b * kL + qg)) * kH + h * 64;
    #pragma unroll
    for (int j = 0; j < 4; ++j)
      Ab[base + j * 16 + l15] = f2bf(accO[j][r] * inv);
  }
}

// ---- kernel 3: output projection (fp32 result) ----------------------------

__global__ __launch_bounds__(256) void out_proj_kernel(
    const __bf16* __restrict__ A, const __bf16* __restrict__ Wo,
    float* __restrict__ out) {
  const int lane = threadIdx.x & 31;
  const int wave = threadIdx.x >> 5;
  const int m0 = blockIdx.x * 128 + (wave & 3) * 32;
  const int n0 = blockIdx.y * 128 + (wave >> 2) * 64;
  v8f acc[2][4];
  gemm_mainloop(A, Wo, m0, n0, lane, acc);
  const int l15 = lane & 15, hi = (lane >> 4) & 1;
  #pragma unroll
  for (int i = 0; i < 2; ++i)
    #pragma unroll
    for (int j = 0; j < 4; ++j)
      #pragma unroll
      for (int r = 0; r < 8; ++r) {
        const int m = m0 + i * 16 + hi * 8 + r;
        const int n = n0 + j * 16 + l15;
        out[(size_t)m * kH + n] = acc[i][j][r];
      }
}

// ---- launch ---------------------------------------------------------------

extern "C" void kernel_launch(void* const* d_in, const int* in_sizes, int n_in,
                              void* d_out, int out_size, void* d_ws, size_t ws_size,
                              hipStream_t stream) {
  (void)in_sizes; (void)n_in; (void)out_size; (void)ws_size;
  const float* X    = (const float*)d_in[0];
  const float* mask = (const float*)d_in[1];
  const float* Wq   = (const float*)d_in[2];
  const float* Wk   = (const float*)d_in[3];
  const float* Wv   = (const float*)d_in[4];
  const float* Wo   = (const float*)d_in[5];
  float* out = (float*)d_out;

  char* ws = (char*)d_ws;
  __bf16* Xb  = (__bf16*)(ws);                     // 16 MB
  __bf16* Wqb = (__bf16*)(ws + 16777216);          //  2 MB
  __bf16* Wkb = (__bf16*)(ws + 18874368);
  __bf16* Wvb = (__bf16*)(ws + 20971520);
  __bf16* Wob = (__bf16*)(ws + 23068672);
  __bf16* Qb  = (__bf16*)(ws + 25165824);          // 16 MB each
  __bf16* Kb  = (__bf16*)(ws + 41943040);
  __bf16* Vb  = (__bf16*)(ws + 58720256);
  __bf16* Ab  = (__bf16*)(ws + 75497472);          // total ~92 MB

  const int nX = kB * kL * kH;   // 8388608
  const int nW = kH * kH;        // 1048576
  cvt_kernel<<<(nX / 4) / 256, 256, 0, stream>>>(X,  Xb,  nX / 4);
  cvt_kernel<<<(nW / 4) / 256, 256, 0, stream>>>(Wq, Wqb, nW / 4);
  cvt_kernel<<<(nW / 4) / 256, 256, 0, stream>>>(Wk, Wkb, nW / 4);
  cvt_kernel<<<(nW / 4) / 256, 256, 0, stream>>>(Wv, Wvb, nW / 4);
  cvt_kernel<<<(nW / 4) / 256, 256, 0, stream>>>(Wo, Wob, nW / 4);

  dim3 g1((kB * kL) / 128, kH / 128, 3);
  qkv_proj_kernel<<<g1, 256, 0, stream>>>(Xb, Wqb, Wkb, Wvb, Qb, Kb, Vb);

  dim3 g2(kL / 64, kB * kNH);
  flash_kernel<<<g2, 128, 0, stream>>>(Qb, Kb, Vb, mask, Ab);

  dim3 g3((kB * kL) / 128, kH / 128);
  out_proj_kernel<<<g3, 256, 0, stream>>>(Ab, Wob, out);
}
